// MultiHeadSelfAttention_39848706573457
// MI455X (gfx1250) — matrix operand order repaired and checked by reading, not compiled, not run on hardware
//
#include <hip/hip_runtime.h>
#include <hip/hip_bf16.h>
#include <cstdint>

// ---------------------------------------------------------------------------
// Multi-head self-attention, B=8 T=1024 D=768 H=12 hd=64, fp32 in/out.
// Compute-bound (~65 GFLOP vs ~7us of HBM traffic) -> everything through
// v_wmma_f32_16x16x32_bf16. All WMMA operands are pre-laid-out bf16 in
// global memory so fragments are two contiguous 16B loads per lane
// (matching CDNA5 ISA 7.12.2 layouts) -> LDS-free, barrier-free
// GEMM hot loops that the compiler can software-pipeline (round-1 evidence:
// global_load_b128 directly feeding v_wmma under s_clause).
// ---------------------------------------------------------------------------

typedef __attribute__((ext_vector_type(16))) __bf16 bf16x16;
typedef __attribute__((ext_vector_type(8)))  float  f32x8;

union Frag {
    bf16x16 v;
    uint4 q[2];
    unsigned short u[16];
};

__device__ __forceinline__ unsigned short f2bf(float f) {
    unsigned int u = __float_as_uint(f);
    u += 0x7FFFu + ((u >> 16) & 1u);     // round-to-nearest-even
    return (unsigned short)(u >> 16);
}

__device__ __forceinline__ f32x8 wmma_bf16(const Frag& a, const Frag& b, f32x8 c) {
    return __builtin_amdgcn_wmma_f32_16x16x32_bf16(
        /*neg_a=*/false, a.v, /*neg_b=*/false, b.v,
        /*c_mod=*/(short)0, c, /*reuse_a=*/false, /*reuse_b=*/false);
}

// ---------------------------------------------------------------------------
// Pre-pass 1: fp32 -> bf16 elementwise (for x). Bandwidth-bound.
// ---------------------------------------------------------------------------
__global__ __launch_bounds__(256)
void f32_to_bf16_kernel(const float* __restrict__ src,
                        unsigned short* __restrict__ dst)
{
    size_t i = ((size_t)blockIdx.x * 256 + threadIdx.x) * 4;
    float4 v = *(const float4*)(src + i);
    ushort4 o;
    o.x = f2bf(v.x); o.y = f2bf(v.y); o.z = f2bf(v.z); o.w = f2bf(v.w);
    *(ushort4*)(dst + i) = o;
}

// ---------------------------------------------------------------------------
// Pre-pass 2: W[K][N] fp32 -> Wt[N][K] bf16 (32x32 LDS tile transpose).
// ---------------------------------------------------------------------------
__global__ __launch_bounds__(256)
void transpose_to_bf16(const float* __restrict__ W,
                       unsigned short* __restrict__ Wt, int K, int N)
{
    __shared__ unsigned short t[32][33];
    const int n0 = blockIdx.x * 32, k0 = blockIdx.y * 32;
    const int tx = threadIdx.x & 31, ty = threadIdx.x >> 5;   // 32 x 8
    #pragma unroll
    for (int i = 0; i < 32; i += 8)
        t[ty + i][tx] = f2bf(W[(size_t)(k0 + ty + i) * N + n0 + tx]);
    __syncthreads();
    #pragma unroll
    for (int i = 0; i < 32; i += 8)
        Wt[(size_t)(n0 + ty + i) * K + k0 + tx] = t[tx][ty + i];
}

// ---------------------------------------------------------------------------
// LDS-free GEMM: C[M,N] = Abf[M,K] * Btb[N,K]^T (+bias).
// 4 waves/block (2x2), each wave owns a 64x64 tile = 4x4 f32x8 accumulators.
// Per 32-wide k-step: 16 contiguous b128 loads, 16 WMMAs, zero barriers.
// Weights (bf16, <=3.5MB) stay L2-resident across the 64 M-block-rows.
// ---------------------------------------------------------------------------
template<bool QKV_EPI>
__global__ __launch_bounds__(128)
void gemm_wmma_reg(const unsigned short* __restrict__ Abf,   // [M][K] bf16
                   const unsigned short* __restrict__ Btb,   // [N][K] bf16
                   const float* __restrict__ bias,
                   int M, int N, int K,
                   float* __restrict__ outF,
                   unsigned short* __restrict__ Qo,
                   unsigned short* __restrict__ Ko,
                   unsigned short* __restrict__ Vto)
{
    const int lane  = threadIdx.x & 31;
    const int wave  = threadIdx.x >> 5;    // 0..3
    const int waveM = wave >> 1;
    const int waveN = wave & 1;
    const int m0 = blockIdx.y * 128 + waveM * 64;
    const int n0 = blockIdx.x * 128 + waveN * 64;

    const int kb   = (lane >> 4) * 8;     // A-frag K sub-chunk per lane-half
    const int koff = (lane >> 4) * 16;    // unused; B-frag also uses kb
    const int nlo  = lane & 15;

    const f32x8 zero = {0.f,0.f,0.f,0.f,0.f,0.f,0.f,0.f};
    f32x8 acc[4][4];
    #pragma unroll
    for (int i = 0; i < 4; i++)
        #pragma unroll
        for (int j = 0; j < 4; j++)
            acc[i][j] = zero;

    const unsigned short* aRow[4];
    const unsigned short* bRow[4];
    #pragma unroll
    for (int sm = 0; sm < 4; sm++)
        aRow[sm] = Abf + (size_t)(m0 + sm * 16 + nlo) * K;
    #pragma unroll
    for (int sn = 0; sn < 4; sn++)
        bRow[sn] = Btb + (size_t)(n0 + sn * 16 + nlo) * K;

    for (int k0 = 0; k0 < K; k0 += 32) {
        Frag a[4];
        #pragma unroll
        for (int sm = 0; sm < 4; sm++) {
            a[sm].q[0] = *(const uint4*)(aRow[sm] + k0 + kb);
            a[sm].q[1] = *(const uint4*)(aRow[sm] + k0 + 16 + kb);
        }
        #pragma unroll
        for (int sn = 0; sn < 4; sn++) {
            Frag b;
            b.q[0] = *(const uint4*)(bRow[sn] + k0 + kb);
            b.q[1] = *(const uint4*)(bRow[sn] + k0 + 16 + kb);
            #pragma unroll
            for (int sm = 0; sm < 4; sm++)
                acc[sm][sn] = wmma_bf16(a[sm], b, acc[sm][sn]);
        }
    }

    // ---- epilogue ----
    const int rbase = 8 * (lane >> 4);
    #pragma unroll
    for (int sm = 0; sm < 4; sm++) {
        #pragma unroll
        for (int sn = 0; sn < 4; sn++) {
            int col = n0 + sn * 16 + nlo;
            float bv = bias[col];
            #pragma unroll
            for (int r = 0; r < 8; r++) {
                int row = m0 + sm * 16 + rbase + r;
                float val = acc[sm][sn][r] + bv;
                if constexpr (QKV_EPI) {
                    int s = (col >= 1536) ? 2 : ((col >= 768) ? 1 : 0);
                    int rem = col - s * 768;
                    int h = rem >> 6, d = rem & 63;
                    int b_ = row >> 10, t = row & 1023;   // T = 1024
                    size_t bh = (size_t)(b_ * 12 + h);
                    unsigned short bfv = f2bf(val);
                    if (s == 0)      Qo[(bh * 1024 + t) * 64 + d] = bfv;
                    else if (s == 1) Ko[(bh * 1024 + t) * 64 + d] = bfv;
                    else             Vto[(bh * 64 + d) * 1024 + t] = bfv;
                } else {
                    outF[(size_t)row * N + col] = val;
                }
            }
        }
    }
}

// ---------------------------------------------------------------------------
// Flash attention: one wave = 16 queries. Online softmax over 32-key tiles:
// 4 WMMAs for S = Q*K^T, cross-lane row max/sum via shfl_xor within 16-lane
// halves, P -> A-frag via per-wave private LDS bounce (LDS in-order per
// wave, no barrier), 4 WMMAs for the PV accumulate.
// ---------------------------------------------------------------------------
#define LDP 48   // LDS row pitch in bf16 elems: 96B rows -> 16B-aligned loads

__global__ __launch_bounds__(256)
void attn_flash(const unsigned short* __restrict__ Qbf,
                const unsigned short* __restrict__ Kbf,
                const unsigned short* __restrict__ Vt,
                unsigned short* __restrict__ Ybf)
{
    __shared__ unsigned short Plds[8][16 * LDP];
    const int T = 1024;
    const int lane  = threadIdx.x & 31;
    const int wave  = threadIdx.x >> 5;
    const int bh    = blockIdx.y;          // 0..95
    const int b     = bh / 12;
    const int h     = bh % 12;
    const int qBase = blockIdx.x * 128 + wave * 16;

    const int kb   = (lane >> 4) * 8;
    const int koff = (lane >> 4) * 16;
    const int nlo  = lane & 15;

    // Q rows for this wave: two 16x32 A-fragments, kept in registers.
    Frag aQ[2];
    {
        const unsigned short* qptr = Qbf + ((size_t)bh * T + qBase + nlo) * 64;
        #pragma unroll
        for (int c = 0; c < 2; c++) {
            aQ[c].q[0] = *(const uint4*)(qptr + c * 32 + kb);
            aQ[c].q[1] = *(const uint4*)(qptr + c * 32 + 16 + kb);
        }
    }

    const f32x8 zero = {0.f,0.f,0.f,0.f,0.f,0.f,0.f,0.f};
    f32x8 accY[4];
    #pragma unroll
    for (int f = 0; f < 4; f++) accY[f] = zero;
    float mrow[8], lrow[8];
    #pragma unroll
    for (int r = 0; r < 8; r++) { mrow[r] = -3.0e38f; lrow[r] = 0.f; }

    for (int kt = 0; kt < T / 32; kt++) {
        // ---- S = Q * K^T for 32 keys ----
        f32x8 s[2] = {zero, zero};
        #pragma unroll
        for (int j = 0; j < 2; j++) {
            const unsigned short* kp =
                Kbf + ((size_t)bh * T + kt * 32 + j * 16 + nlo) * 64;
            #pragma unroll
            for (int c = 0; c < 2; c++) {
                Frag bK;
                bK.q[0] = *(const uint4*)(kp + c * 32 + kb);
                bK.q[1] = *(const uint4*)(kp + c * 32 + 16 + kb);
                s[j] = wmma_bf16(aQ[c], bK, s[j]);
            }
        }

        // ---- online softmax (per row r in the C-layout) ----
        const float scale = 0.125f;   // 1/sqrt(64)
        float p0[8], p1[8];
        #pragma unroll
        for (int r = 0; r < 8; r++) {
            float s0 = s[0][r] * scale;
            float s1 = s[1][r] * scale;
            float mx = fmaxf(s0, s1);
            mx = fmaxf(mx, __shfl_xor(mx, 1));
            mx = fmaxf(mx, __shfl_xor(mx, 2));
            mx = fmaxf(mx, __shfl_xor(mx, 4));
            mx = fmaxf(mx, __shfl_xor(mx, 8));
            float mn = fmaxf(mrow[r], mx);
            float alpha = __expf(mrow[r] - mn);
            mrow[r] = mn;
            p0[r] = __expf(s0 - mn);
            p1[r] = __expf(s1 - mn);
            float rs = p0[r] + p1[r];
            rs += __shfl_xor(rs, 1);
            rs += __shfl_xor(rs, 2);
            rs += __shfl_xor(rs, 4);
            rs += __shfl_xor(rs, 8);
            lrow[r] = lrow[r] * alpha + rs;
            #pragma unroll
            for (int f = 0; f < 4; f++) accY[f][r] *= alpha;
        }

        // ---- P (C-layout) -> A-fragment via per-wave LDS bounce ----
        unsigned short* pl = &Plds[wave][0];
        const int rbase = 8 * (lane >> 4);
        #pragma unroll
        for (int r = 0; r < 8; r++) {
            pl[(rbase + r) * LDP + nlo]      = f2bf(p0[r]);
            pl[(rbase + r) * LDP + 16 + nlo] = f2bf(p1[r]);
        }
        Frag aP;
        aP.q[0] = *(const uint4*)&pl[nlo * LDP + kb];
        aP.q[1] = *(const uint4*)&pl[nlo * LDP + 16 + kb];

        // ---- accY += P * V  (V stored transposed [bh][d][t]) ----
        #pragma unroll
        for (int f = 0; f < 4; f++) {
            const unsigned short* vp =
                Vt + ((size_t)bh * 64 + f * 16 + nlo) * T + kt * 32 + kb;
            Frag bV;
            bV.q[0] = *(const uint4*)vp;
            bV.q[1] = *(const uint4*)(vp + 16);
            accY[f] = wmma_bf16(aP, bV, accY[f]);
        }
    }

    // ---- normalize and store Y as bf16 [B, T, 768] for the proj GEMM ----
    const int rbase = 8 * (lane >> 4);
    #pragma unroll
    for (int r = 0; r < 8; r++) {
        float inv = 1.0f / lrow[r];
        int t = qBase + rbase + r;
        size_t base = ((size_t)b * T + t) * 768 + h * 64;
        #pragma unroll
        for (int f = 0; f < 4; f++)
            Ybf[base + f * 16 + nlo] = f2bf(accY[f][r] * inv);
    }
}

// ---------------------------------------------------------------------------
extern "C" void kernel_launch(void* const* d_in, const int* in_sizes, int n_in,
                              void* d_out, int out_size, void* d_ws, size_t ws_size,
                              hipStream_t stream)
{
    const float* x      = (const float*)d_in[0];   // [8,1024,768]
    const float* W_qkv  = (const float*)d_in[1];   // [768,2304]
    const float* b_qkv  = (const float*)d_in[2];   // [2304]
    const float* W_proj = (const float*)d_in[3];   // [768,768]
    const float* b_proj = (const float*)d_in[4];   // [768]
    float* out = (float*)d_out;                    // [8,1024,768]
    (void)ws_size; (void)in_sizes; (void)n_in; (void)out_size;

    const int B = 8, T = 1024, D = 768, H = 12;
    const int M = B * T;                           // 8192
    const size_t perBH = (size_t)B * H * T * 64;   // 6291456 elems

    // workspace layout (bf16 = unsigned short), ~68 MB total
    unsigned short* Qbf   = (unsigned short*)d_ws;       // [B,H,T,64]
    unsigned short* Kbf   = Qbf   + perBH;               // [B,H,T,64]
    unsigned short* Vt    = Kbf   + perBH;               // [B,H,64,T]
    unsigned short* Ybf   = Vt    + perBH;               // [B,T,768]
    unsigned short* xbf   = Ybf   + perBH;               // [M,768]
    unsigned short* Wqkvt = xbf   + (size_t)M * D;       // [2304,768]
    unsigned short* Wprjt = Wqkvt + (size_t)3 * D * D;   // [768,768]

    // pre-passes: bf16 conversion + weight transposes (bandwidth-trivial)
    f32_to_bf16_kernel<<<(M * D) / 1024, 256, 0, stream>>>(x, xbf);
    transpose_to_bf16<<<dim3((3 * D) / 32, D / 32), 256, 0, stream>>>(
        W_qkv, Wqkvt, D, 3 * D);
    transpose_to_bf16<<<dim3(D / 32, D / 32), 256, 0, stream>>>(
        W_proj, Wprjt, D, D);

    // 1) QKV projection + bias + scatter to attention layouts
    gemm_wmma_reg<true><<<dim3((3 * D) / 128, M / 128), 128, 0, stream>>>(
        xbf, Wqkvt, b_qkv, M, 3 * D, D, nullptr, Qbf, Kbf, Vt);

    // 2) flash attention (96 bh-slices x 8 query-blocks of 128)
    attn_flash<<<dim3(T / 128, B * H), 256, 0, stream>>>(Qbf, Kbf, Vt, Ybf);

    // 3) output projection + bias -> fp32
    gemm_wmma_reg<false><<<dim3(D / 128, M / 128), 128, 0, stream>>>(
        Ybf, Wprjt, b_proj, M, D, D, out, nullptr, nullptr, nullptr);
}
